// MidBlockWithCrossAttn_10668698764060
// MI455X (gfx1250) — matrix operand order repaired and checked by reading, not compiled, not run on hardware
//
#include <hip/hip_runtime.h>

typedef __bf16 bf16;
typedef __attribute__((ext_vector_type(8)))  __bf16 bf16x8;
typedef __attribute__((ext_vector_type(16))) __bf16 v16bf;
typedef __attribute__((ext_vector_type(8)))  float  v8f;

// ---------------------------------------------------------------- device utils
__device__ __forceinline__ bf16x8 ldg8(const bf16* p) {
  return *reinterpret_cast<const bf16x8*>(p);
}
__device__ __forceinline__ bf16x8 bzero8() {
  bf16x8 z;
#pragma unroll
  for (int j = 0; j < 8; ++j) z[j] = (bf16)0.0f;
  return z;
}
// Assemble the 16-element per-lane operand from two 8-element contiguous loads.
__device__ __forceinline__ v16bf mk16(bf16x8 lo, bf16x8 hi) {
  v16bf r;
#pragma unroll
  for (int j = 0; j < 8; ++j) { r[j] = lo[j]; r[j + 8] = hi[j]; }
  return r;
}
__device__ __forceinline__ v8f wmma_bf16(v16bf a, v16bf b, v8f c) {
  return __builtin_amdgcn_wmma_f32_16x16x32_bf16(false, a, false, b, (short)0, c,
                                                 false, false);
}
__device__ __forceinline__ float silu_f(float x) { return x / (1.0f + expf(-x)); }

// ---------------------------------------------------------------- WMMA GEMM
// A: bf16 [Mtot,K] row-major.  W: bf16 [Ntot,K] row-major (i.e. B[k][n]=W[n][k]).
// out = act(A*W^T + bias) + res ; written f32 and/or bf16, row-major [Mtot,Ntot].
// Wave computes a 32x64 tile: 2 M-subtiles x 4 N-subtiles (8 accumulators);
// each B fragment is reused for two WMMAs -> ~21 FLOP/byte from cache.
// A-operand lane layout (16-bit A 16x32): lane<16 holds K {0..7,16..23},
// lane>=16 holds K {8..15,24..31} of row (lane&15).
// B-operand: lane holds col (lane&15); same K order as A.
template <bool FULL>
__device__ __forceinline__ void gemm_core(
    const bf16* __restrict__ A, const bf16* __restrict__ W,
    const float* __restrict__ bias, const float* res,
    float* outf, bf16* outb, int Mtot, int Ntot, int K, int act,
    int m0, int n0, int lane) {
  const int r15 = lane & 15;
  const int hi  = lane >> 4;
  const bf16* ap0 = A + (size_t)(m0 + r15) * K + hi * 8;
  const bf16* ap1 = ap0 + (size_t)16 * K;
  bool av0 = true, av1 = true;
  if (!FULL) { av0 = (m0 + r15) < Mtot; av1 = (m0 + 16 + r15) < Mtot; }
  const bf16* bp[4];
#pragma unroll
  for (int t = 0; t < 4; ++t)
    bp[t] = W + (size_t)(n0 + t * 16 + r15) * K + hi * 8;
  __builtin_prefetch(ap0, 0, 3);
  __builtin_prefetch(bp[0], 0, 3);
  v8f acc[2][4] = {};
  const bf16x8 zz = bzero8();
#pragma unroll 2
  for (int k0 = 0; k0 < K; k0 += 32) {
    v16bf a0, a1;
    if (FULL) {
      a0 = mk16(ldg8(ap0 + k0), ldg8(ap0 + k0 + 16));
      a1 = mk16(ldg8(ap1 + k0), ldg8(ap1 + k0 + 16));
    } else {
      a0 = av0 ? mk16(ldg8(ap0 + k0), ldg8(ap0 + k0 + 16)) : mk16(zz, zz);
      a1 = av1 ? mk16(ldg8(ap1 + k0), ldg8(ap1 + k0 + 16)) : mk16(zz, zz);
    }
#pragma unroll
    for (int t = 0; t < 4; ++t) {
      v16bf b = mk16(ldg8(bp[t] + k0), ldg8(bp[t] + k0 + 16));
      acc[0][t] = wmma_bf16(a0, b, acc[0][t]);
      acc[1][t] = wmma_bf16(a1, b, acc[1][t]);
    }
  }
#pragma unroll
  for (int mt = 0; mt < 2; ++mt) {
#pragma unroll
    for (int t = 0; t < 4; ++t) {
      const int col = n0 + t * 16 + r15;
      const float bz = bias ? bias[col] : 0.0f;
#pragma unroll
      for (int r = 0; r < 8; ++r) {
        const int m = m0 + mt * 16 + r + hi * 8;  // D: lanes 0-15 M=r, 16-31 M=8+r
        if (FULL || m < Mtot) {
          float v = acc[mt][t][r] + bz;
          if (act == 1) v = fmaxf(v, 0.0f);
          else if (act == 2) v = 0.5f * v * (1.0f + erff(v * 0.70710678118654752f));
          const size_t o = (size_t)m * Ntot + col;
          if (res)  v += res[o];
          if (outf) outf[o] = v;
          if (outb) outb[o] = (bf16)v;
        }
      }
    }
  }
}

__global__ __launch_bounds__(128) void mb_gemm(
    const bf16* __restrict__ A, const bf16* __restrict__ W,
    const float* __restrict__ bias, const float* res,
    float* outf, bf16* outb, int Mtot, int Ntot, int K, int act) {
  const int lane = threadIdx.x & 31;
  const int wid  = threadIdx.x >> 5;
  const int m0   = (blockIdx.x * 4 + wid) << 5;   // 32 rows per wave
  if (m0 >= Mtot) return;
  const int n0 = blockIdx.y << 6;
  if (m0 + 32 <= Mtot)
    gemm_core<true>(A, W, bias, res, outf, outb, Mtot, Ntot, K, act, m0, n0, lane);
  else
    gemm_core<false>(A, W, bias, res, outf, outb, Mtot, Ntot, K, act, m0, n0, lane);
}

// ---------------------------------------------------------------- 3x3 conv (implicit GEMM)
// A: bf16 tok [8192,512]; W3: bf16 [512][9][512] (o, ky*3+kx, cin).
// out f32 tok [8192,512] = conv + bias (+ res).
// Wave computes 32 tokens x 64 out-channels (one full image row per wave).
__global__ __launch_bounds__(128) void mb_conv3(
    const bf16* __restrict__ A, const bf16* __restrict__ W3,
    const float* __restrict__ bias, const float* res, float* outf) {
  const int lane = threadIdx.x & 31;
  const int wid  = threadIdx.x >> 5;
  const int m0   = (blockIdx.x * 4 + wid) << 5;   // grid.x = 64 -> 8192 tokens
  const int n0   = blockIdx.y << 6;               // grid.y = 8  -> 512 out ch
  const int r15  = lane & 15;
  const int hi   = lane >> 4;
  const int b    = m0 >> 10;
  const int nl   = m0 & 1023;                     // multiple of 32 -> one row
  const int y    = nl >> 5;
  v8f acc[2][4] = {};
  const bf16x8 zz = bzero8();
  for (int ky = 0; ky < 3; ++ky) {
    const int yy = y + ky - 1;
    if ((unsigned)yy >= 32u) continue;
    const size_t rowbase = (size_t)((b << 10) + (yy << 5));
    for (int kx = 0; kx < 3; ++kx) {
      const int x0 = r15 + kx - 1;        // tile 0: x = 0..15
      const int x1 = x0 + 16;             // tile 1: x = 16..31
      const bool v0 = (unsigned)x0 < 32u;
      const bool v1 = (unsigned)x1 < 32u;
      const bf16* ap0 = A + (rowbase + x0) * 512 + hi * 8;
      const bf16* ap1 = A + (rowbase + x1) * 512 + hi * 8;
      const int kyx = ky * 3 + kx;
      const bf16* wp[4];
#pragma unroll
      for (int t = 0; t < 4; ++t)
        wp[t] = W3 + (size_t)(n0 + t * 16 + r15) * 4608 + kyx * 512 + hi * 8;
#pragma unroll 2
      for (int c0 = 0; c0 < 512; c0 += 32) {
        v16bf a0 = v0 ? mk16(ldg8(ap0 + c0), ldg8(ap0 + c0 + 16)) : mk16(zz, zz);
        v16bf a1 = v1 ? mk16(ldg8(ap1 + c0), ldg8(ap1 + c0 + 16)) : mk16(zz, zz);
#pragma unroll
        for (int t = 0; t < 4; ++t) {
          v16bf bb = mk16(ldg8(wp[t] + c0), ldg8(wp[t] + c0 + 16));
          acc[0][t] = wmma_bf16(a0, bb, acc[0][t]);
          acc[1][t] = wmma_bf16(a1, bb, acc[1][t]);
        }
      }
    }
  }
#pragma unroll
  for (int mt = 0; mt < 2; ++mt) {
#pragma unroll
    for (int t = 0; t < 4; ++t) {
      const int col = n0 + t * 16 + r15;
      const float bz = bias[col];
#pragma unroll
      for (int r = 0; r < 8; ++r) {
        const int m = m0 + mt * 16 + r + hi * 8;
        float v = acc[mt][t][r] + bz;
        const size_t o = (size_t)m * 512 + col;
        if (res) v += res[o];
        outf[o] = v;
      }
    }
  }
}

// ---------------------------------------------------------------- small kernels
__global__ void mb_cvtbf(const float* __restrict__ in, bf16* __restrict__ out, int n) {
  int i = blockIdx.x * 256 + threadIdx.x;
  if (i < n) out[i] = (bf16)in[i];
}

// OIHW f32 [512,512,3,3] -> bf16 [512][9][512]
__global__ void mb_convw(const float* __restrict__ w, bf16* __restrict__ out) {
  int i = blockIdx.x * 256 + threadIdx.x;
  if (i >= 512 * 4608) return;
  int o = i / 4608, rem = i - o * 4608;
  int kyx = rem >> 9, ci = rem & 511;
  out[i] = (bf16)w[(size_t)o * 4608 + ci * 9 + kyx];
}

// NCHW f32 -> tok f32  (tok[(b*1024+n)*512+c] = x[b][c][n])
__global__ void mb_nchw2tok(const float* __restrict__ x, float* __restrict__ tok) {
  int idx = blockIdx.x * 256 + threadIdx.x;      // 8*1024*512 = 2^22
  int c = idx & 511, m = idx >> 9, b = m >> 10, n = m & 1023;
  tok[idx] = x[((size_t)b << 19) + ((size_t)c << 10) + n];
}
__global__ void mb_tok2nchw(const float* __restrict__ tok, float* __restrict__ out) {
  int idx = blockIdx.x * 256 + threadIdx.x;
  int n = idx & 1023, c = (idx >> 10) & 511, b = idx >> 19;
  out[idx] = tok[((size_t)((b << 10) + n)) * 512 + c];
}

// GroupNorm stats: 64 blocks (b,g). stats[bg*2]=mean, [bg*2+1]=rstd over 64ch*1024px.
__global__ void mb_gnstats(const float* __restrict__ h, float* __restrict__ stats) {
  int bg = blockIdx.x, b = bg >> 3, g = bg & 7;
  __shared__ float rs[256], rq[256];
  float s = 0.f, q = 0.f;
  for (int i = threadIdx.x; i < 65536; i += 256) {
    int c = (g << 6) + (i & 63);
    int n = i >> 6;
    float v = h[((size_t)((b << 10) + n)) * 512 + c];
    s += v; q += v * v;
  }
  rs[threadIdx.x] = s; rq[threadIdx.x] = q;
  __syncthreads();
  for (int off = 128; off; off >>= 1) {
    if ((int)threadIdx.x < off) {
      rs[threadIdx.x] += rs[threadIdx.x + off];
      rq[threadIdx.x] += rq[threadIdx.x + off];
    }
    __syncthreads();
  }
  if (threadIdx.x == 0) {
    float mean = rs[0] * (1.0f / 65536.0f);
    float var  = rq[0] * (1.0f / 65536.0f) - mean * mean;
    stats[bg * 2] = mean;
    stats[bg * 2 + 1] = rsqrtf(var + 1e-5f);
  }
}

// normalize (+optional AdaGN scale/shift) + SiLU -> bf16 tok
__global__ void mb_gnapply(const float* __restrict__ h, const float* __restrict__ stats,
                           const float* __restrict__ gg, const float* __restrict__ gb,
                           const float* __restrict__ tvec, bf16* __restrict__ outb) {
  int idx = blockIdx.x * 256 + threadIdx.x;     // 8192*512
  int c = idx & 511, m = idx >> 9, b = m >> 10;
  int bg = (b << 3) + (c >> 6);
  float v = (h[idx] - stats[bg * 2]) * stats[bg * 2 + 1] * gg[c] + gb[c];
  if (tvec) v = v * (1.0f + tvec[(b << 10) + c]) + tvec[(b << 10) + 512 + c];
  outb[idx] = (bf16)silu_f(v);
}

// LayerNorm over 512 channels per token; optional f32 and/or bf16 outputs.
__global__ void mb_ln(const float* __restrict__ x, const float* __restrict__ g,
                      const float* __restrict__ bb, float* outf, bf16* outb) {
  int m = blockIdx.x;
  const float* row = x + (size_t)m * 512;
  __shared__ float rs[256], rq[256];
  float v0 = row[threadIdx.x], v1 = row[threadIdx.x + 256];
  rs[threadIdx.x] = v0 + v1;
  rq[threadIdx.x] = v0 * v0 + v1 * v1;
  __syncthreads();
  for (int off = 128; off; off >>= 1) {
    if ((int)threadIdx.x < off) {
      rs[threadIdx.x] += rs[threadIdx.x + off];
      rq[threadIdx.x] += rq[threadIdx.x + off];
    }
    __syncthreads();
  }
  float mean = rs[0] * (1.0f / 512.0f);
  float rstd = rsqrtf(rq[0] * (1.0f / 512.0f) - mean * mean + 1e-5f);
#pragma unroll
  for (int p = 0; p < 2; ++p) {
    int i = threadIdx.x + p * 256;
    float y = (row[i] - mean) * rstd * g[i] + bb[i];
    size_t o = (size_t)m * 512 + i;
    if (outf) outf[o] = y;
    if (outb) outb[o] = (bf16)y;
  }
}

// t = silu(t_emb) @ time_w^T + time_b  -> tvec [8,1024]
__global__ void mb_time(const float* __restrict__ te, const float* __restrict__ tw,
                        const float* __restrict__ tb, float* __restrict__ tvec) {
  int idx = blockIdx.x * 256 + threadIdx.x;     // 8192
  int b = idx >> 10, j = idx & 1023;
  const float* tr = te + b * 256;
  const float* wr = tw + (size_t)j * 256;
  float acc = tb[j];
  for (int k = 0; k < 256; ++k) acc += silu_f(tr[k]) * wr[k];
  tvec[idx] = acc;
}

// SA quirk scores: sc[b,h,d,e] = sum_n q[n,32h+d]/sqrt32 * k[n,32h+e]; qkv tok [8192,1536]
__global__ void mb_sascores(const float* __restrict__ qkv, float* __restrict__ sc) {
  int bh = blockIdx.x, b = bh >> 4, hh = bh & 15;
  __shared__ float qs[128][33];
  __shared__ float ks[128][33];
  float acc[4] = {0.f, 0.f, 0.f, 0.f};
  for (int n0 = 0; n0 < 1024; n0 += 128) {
    __syncthreads();
    for (int t = threadIdx.x; t < 128 * 32; t += 256) {
      int j = t >> 5, e = t & 31;
      size_t rb = ((size_t)((b << 10) + n0 + j)) * 1536 + hh * 32 + e;
      qs[j][e] = qkv[rb] * 0.17677669529663687f;   // 1/sqrt(32)
      ks[j][e] = qkv[rb + 512];
    }
    __syncthreads();
#pragma unroll
    for (int p4 = 0; p4 < 4; ++p4) {
      int p = threadIdx.x + p4 * 256;
      int d = p >> 5, e = p & 31;
      float a = acc[p4];
      for (int j = 0; j < 128; ++j) a += qs[j][d] * ks[j][e];
      acc[p4] = a;
    }
  }
#pragma unroll
  for (int p4 = 0; p4 < 4; ++p4)
    sc[(size_t)bh * 1024 + threadIdx.x + p4 * 256] = acc[p4];
}

__global__ void mb_softmax32(float* sc) {   // 4096 rows of 32; block=32
  size_t base = (size_t)blockIdx.x * 32;
  float v = sc[base + threadIdx.x];
  float m = v;
  for (int off = 16; off; off >>= 1) m = fmaxf(m, __shfl_xor(m, off, 32));
  float e = expf(v - m);
  float s = e;
  for (int off = 16; off; off >>= 1) s += __shfl_xor(s, off, 32);
  sc[base + threadIdx.x] = e / s;
}

// out_tok[m,32h+d] = sum_e attn[b,h,d,e] * v_tok[m,1024+32h+e]  -> bf16
__global__ void mb_saattn(const float* __restrict__ qkv, const float* __restrict__ attn,
                          bf16* __restrict__ outb) {
  int idx = blockIdx.x * 256 + threadIdx.x;     // 8192*512
  int c = idx & 511, m = idx >> 9, b = m >> 10;
  int hh = c >> 5, d = c & 31;
  const float* ar = attn + ((size_t)(((b << 4) + hh) * 32 + d)) * 32;
  const float* vr = qkv + (size_t)m * 1536 + 1024 + hh * 32;
  float acc = 0.f;
#pragma unroll
  for (int e = 0; e < 32; ++e) acc += ar[e] * vr[e];
  outb[idx] = (bf16)acc;
}

// CA quirk: q_in[b,m,c] = feat_flat[b, c*1024 + m]   -> bf16
__global__ void mb_quirkq(const float* __restrict__ feat, bf16* __restrict__ outb) {
  int idx = blockIdx.x * 256 + threadIdx.x;     // 8*1024*512
  int b = idx >> 19, j = idx & ((1 << 19) - 1);
  int c = j & 511, mloc = j >> 9;
  outb[idx] = (bf16)feat[((size_t)b << 19) + ((size_t)c << 10) + mloc];
}

// sc[b,h,n,l] = sum_e q[m,32h+e]*k[(b*77+l),32h+e] / sqrt(32)
__global__ void mb_cascores(const float* __restrict__ qf, const float* __restrict__ kf,
                            float* __restrict__ sc, int total) {
  int idx = blockIdx.x * 256 + threadIdx.x;
  if (idx >= total) return;
  int l = idx % 77, r = idx / 77;
  int n = r & 1023, bh = r >> 10, b = bh >> 4, hh = bh & 15;
  const float* qr = qf + ((size_t)((b << 10) + n)) * 512 + hh * 32;
  const float* kr = kf + ((size_t)(b * 77 + l)) * 512 + hh * 32;
  float acc = 0.f;
#pragma unroll
  for (int e = 0; e < 32; ++e) acc += qr[e] * kr[e];
  sc[idx] = acc * 0.17677669529663687f;
}

__global__ void mb_softmax77(float* sc) {   // 131072 rows of 77; block=32
  size_t base = (size_t)blockIdx.x * 77;
  int t = threadIdx.x;
  float v0 = sc[base + t];
  float v1 = (t + 32 < 77) ? sc[base + t + 32] : -1e30f;
  float v2 = (t + 64 < 77) ? sc[base + t + 64] : -1e30f;
  float m = fmaxf(v0, fmaxf(v1, v2));
  for (int off = 16; off; off >>= 1) m = fmaxf(m, __shfl_xor(m, off, 32));
  float e0 = expf(v0 - m);
  float e1 = (t + 32 < 77) ? expf(v1 - m) : 0.f;
  float e2 = (t + 64 < 77) ? expf(v2 - m) : 0.f;
  float s = e0 + e1 + e2;
  for (int off = 16; off; off >>= 1) s += __shfl_xor(s, off, 32);
  float inv = 1.0f / s;
  sc[base + t] = e0 * inv;
  if (t + 32 < 77) sc[base + t + 32] = e1 * inv;
  if (t + 64 < 77) sc[base + t + 64] = e2 * inv;
}

// out_tok[m,32h+d] = sum_l attn[b,h,n,l] * v[(b*77+l),32h+d]  -> bf16
__global__ void mb_caattn(const float* __restrict__ attn, const float* __restrict__ vf,
                          bf16* __restrict__ outb) {
  int idx = blockIdx.x * 256 + threadIdx.x;     // 8192*512
  int c = idx & 511, m = idx >> 9, b = m >> 10, n = m & 1023;
  int hh = c >> 5, d = c & 31;
  const float* ar = attn + ((size_t)(((b << 4) + hh) * 1024 + n)) * 77;
  const float* vr = vf + (size_t)(b * 77) * 512 + hh * 32 + d;
  float acc = 0.f;
  for (int l = 0; l < 77; ++l) acc += ar[l] * vr[(size_t)l * 512];
  outb[idx] = (bf16)acc;
}

// ---------------------------------------------------------------- host side
struct ResW { const float *gn1g,*gn1b,*c1b,*tw,*tb,*gn2g,*gn2b,*c2b; bf16 *w1,*w2; };
struct SAW  { const float *ln1g,*ln1b,*qkvb,*projb,*ln2g,*ln2b,*f1b,*f2b;
              bf16 *qkv,*proj,*f1,*f2; };
struct CAW  { const float *lnfg,*lnfb,*lncg,*lncb,*qb,*kb,*vb,*ob,*lnffng,*lnffnb,*f1b,*f2b;
              bf16 *qw,*kw,*vw,*ow,*f1,*f2; };

static inline int cdiv(int a, int b) { return (a + b - 1) / b; }

extern "C" void kernel_launch(void* const* d_in, const int* in_sizes, int n_in,
                              void* d_out, int out_size, void* d_ws, size_t ws_size,
                              hipStream_t stream) {
  (void)in_sizes; (void)n_in; (void)out_size; (void)ws_size;
  auto F = [&](int i) -> const float* { return (const float*)d_in[i]; };
  const float* X    = F(0);
  const float* TEMB = F(1);
  const float* COND = F(2);

  char* base = (char*)d_ws;
  size_t off = 0;
  auto alloc  = [&](size_t bytes) -> char* {
    char* p = base + off;
    off += (bytes + 255) & ~(size_t)255;
    return p;
  };
  auto allocb = [&](size_t n) -> bf16*  { return (bf16*)alloc(n * 2); };
  auto allocf = [&](size_t n) -> float* { return (float*)alloc(n * 4); };
  auto cvt = [&](const float* src, bf16* dst, int n) {
    mb_cvtbf<<<cdiv(n, 256), 256, 0, stream>>>(src, dst, n);
  };

  // ---- weights: params flatten as in_res(10) then per layer [sa(12), ca(18), res(10)]
  ResW RW[5]; SAW SW[4]; CAW CW[4];
  int rbase[5];
  rbase[0] = 3;
  for (int l = 0; l < 4; ++l) rbase[l + 1] = 13 + 40 * l + 30;
  for (int i = 0; i < 5; ++i) {
    int p = rbase[i];
    RW[i].gn1g = F(p + 0); RW[i].gn1b = F(p + 1); RW[i].c1b = F(p + 3);
    RW[i].tw   = F(p + 4); RW[i].tb   = F(p + 5);
    RW[i].gn2g = F(p + 6); RW[i].gn2b = F(p + 7); RW[i].c2b = F(p + 9);
    RW[i].w1 = allocb(512 * 9 * 512);
    RW[i].w2 = allocb(512 * 9 * 512);
    mb_convw<<<cdiv(512 * 4608, 256), 256, 0, stream>>>(F(p + 2), RW[i].w1);
    mb_convw<<<cdiv(512 * 4608, 256), 256, 0, stream>>>(F(p + 8), RW[i].w2);
  }
  for (int l = 0; l < 4; ++l) {
    int s = 13 + 40 * l;
    SW[l].ln1g = F(s + 0);  SW[l].ln1b = F(s + 1);
    SW[l].qkvb = F(s + 3);  SW[l].projb = F(s + 5);
    SW[l].ln2g = F(s + 6);  SW[l].ln2b = F(s + 7);
    SW[l].f1b  = F(s + 9);  SW[l].f2b  = F(s + 11);
    SW[l].qkv  = allocb(1536 * 512); cvt(F(s + 2),  SW[l].qkv, 1536 * 512);
    SW[l].proj = allocb(512 * 512);  cvt(F(s + 4),  SW[l].proj, 512 * 512);
    SW[l].f1   = allocb(2048 * 512); cvt(F(s + 8),  SW[l].f1, 2048 * 512);
    SW[l].f2   = allocb(512 * 2048); cvt(F(s + 10), SW[l].f2, 512 * 2048);
    int c = s + 12;
    CW[l].lnfg = F(c + 0);  CW[l].lnfb = F(c + 1);
    CW[l].lncg = F(c + 2);  CW[l].lncb = F(c + 3);
    CW[l].qb = F(c + 5); CW[l].kb = F(c + 7); CW[l].vb = F(c + 9); CW[l].ob = F(c + 11);
    CW[l].lnffng = F(c + 12); CW[l].lnffnb = F(c + 13);
    CW[l].f1b = F(c + 15); CW[l].f2b = F(c + 17);
    CW[l].qw = allocb(512 * 512);  cvt(F(c + 4),  CW[l].qw, 512 * 512);
    CW[l].kw = allocb(512 * 512);  cvt(F(c + 6),  CW[l].kw, 512 * 512);
    CW[l].vw = allocb(512 * 512);  cvt(F(c + 8),  CW[l].vw, 512 * 512);
    CW[l].ow = allocb(512 * 512);  cvt(F(c + 10), CW[l].ow, 512 * 512);
    CW[l].f1 = allocb(2048 * 512); cvt(F(c + 14), CW[l].f1, 2048 * 512);
    CW[l].f2 = allocb(512 * 2048); cvt(F(c + 16), CW[l].f2, 512 * 2048);
  }

  // ---- activation workspace
  float* H    = allocf((size_t)8192 * 512);
  float* X1   = allocf((size_t)8192 * 2048);
  float* X2   = allocf((size_t)8192 * 2048);
  bf16*  S1   = allocb((size_t)8192 * 2048);
  bf16*  S2   = allocb((size_t)8192 * 2048);
  float* FEAT = allocf((size_t)8192 * 512);
  float* KFv  = allocf((size_t)616 * 512);
  float* VFv  = allocf((size_t)616 * 512);
  bf16*  CLN  = allocb((size_t)616 * 512);
  float* TVEC = allocf(8192);
  float* STAT = allocf(128);
  float* SAT  = allocf((size_t)8 * 16 * 32 * 32);

  auto gemm = [&](const bf16* A, const bf16* W, const float* bias, const float* res,
                  float* outf, bf16* outb, int M, int Nt, int K, int act) {
    dim3 grid(cdiv(cdiv(M, 32), 4), Nt / 64);
    mb_gemm<<<grid, 128, 0, stream>>>(A, W, bias, res, outf, outb, M, Nt, K, act);
  };
  auto conv3 = [&](const bf16* A, const bf16* W, const float* bias,
                   const float* res, float* outf) {
    mb_conv3<<<dim3(64, 8), 128, 0, stream>>>(A, W, bias, res, outf);
  };

  auto resblock = [&](const ResW& R) {
    mb_gnstats<<<64, 256, 0, stream>>>(H, STAT);
    mb_gnapply<<<16384, 256, 0, stream>>>(H, STAT, R.gn1g, R.gn1b, nullptr, S1);
    conv3(S1, R.w1, R.c1b, nullptr, X1);
    mb_time<<<32, 256, 0, stream>>>(TEMB, R.tw, R.tb, TVEC);
    mb_gnstats<<<64, 256, 0, stream>>>(X1, STAT);
    mb_gnapply<<<16384, 256, 0, stream>>>(X1, STAT, R.gn2g, R.gn2b, TVEC, S2);
    conv3(S2, R.w2, R.c2b, H, H);                       // h = h + conv2(...)
  };

  auto selfattn = [&](const SAW& W) {
    mb_ln<<<8192, 256, 0, stream>>>(H, W.ln1g, W.ln1b, nullptr, S1);
    gemm(S1, W.qkv, W.qkvb, nullptr, X1, nullptr, 8192, 1536, 512, 0);
    mb_sascores<<<128, 256, 0, stream>>>(X1, SAT);
    mb_softmax32<<<4096, 32, 0, stream>>>(SAT);
    mb_saattn<<<16384, 256, 0, stream>>>(X1, SAT, S2);
    gemm(S2, W.proj, W.projb, nullptr, X2, nullptr, 8192, 512, 512, 0);
    mb_ln<<<8192, 256, 0, stream>>>(X2, W.ln2g, W.ln2b, nullptr, S1);
    gemm(S1, W.f1, W.f1b, nullptr, nullptr, S2, 8192, 2048, 512, 1);  // ReLU
    gemm(S2, W.f2, W.f2b, H, H, nullptr, 8192, 512, 2048, 0);         // + h
  };

  auto crossattn = [&](const CAW& W) {
    mb_ln<<<8192, 256, 0, stream>>>(H, W.lnfg, W.lnfb, FEAT, nullptr);
    mb_quirkq<<<16384, 256, 0, stream>>>(FEAT, S1);     // faithful reinterpretation
    mb_ln<<<616, 256, 0, stream>>>(COND, W.lncg, W.lncb, nullptr, CLN);
    gemm(S1, W.qw, W.qb, nullptr, X1, nullptr, 8192, 512, 512, 0);
    gemm(CLN, W.kw, W.kb, nullptr, KFv, nullptr, 616, 512, 512, 0);
    gemm(CLN, W.vw, W.vb, nullptr, VFv, nullptr, 616, 512, 512, 0);
    mb_cascores<<<cdiv(10092544, 256), 256, 0, stream>>>(X1, KFv, X2, 10092544);
    mb_softmax77<<<131072, 32, 0, stream>>>(X2);
    mb_caattn<<<16384, 256, 0, stream>>>(X2, VFv, S2);
    gemm(S2, W.ow, W.ob, FEAT, X1, nullptr, 8192, 512, 512, 0);       // + feat
    mb_ln<<<8192, 256, 0, stream>>>(X1, W.lnffng, W.lnffnb, nullptr, S1);
    gemm(S1, W.f1, W.f1b, nullptr, nullptr, S2, 8192, 2048, 512, 2);  // GELU
    gemm(S2, W.f2, W.f2b, FEAT, H, nullptr, 8192, 512, 2048, 0);      // + feat
  };

  // ---- forward
  mb_nchw2tok<<<16384, 256, 0, stream>>>(X, H);
  resblock(RW[0]);
  for (int l = 0; l < 4; ++l) {
    selfattn(SW[l]);
    crossattn(CW[l]);
    resblock(RW[l + 1]);
  }
  mb_tok2nchw<<<16384, 256, 0, stream>>>(H, (float*)d_out);
}
